// dmpnn_51616916963720
// MI455X (gfx1250) — compile-verified, matrix operand order checked by reading, not run on hardware
//
#include <hip/hip_runtime.h>
#include <cstdint>
#include <cstddef>

typedef __attribute__((ext_vector_type(4)))  __bf16 bf16x4;
typedef __attribute__((ext_vector_type(8)))  __bf16 bf16x8;
typedef __attribute__((ext_vector_type(16))) __bf16 bf16x16;
typedef __attribute__((ext_vector_type(8)))  float  f32x8;

__device__ __forceinline__ float leakyf(float x) { return x >= 0.f ? x : 0.01f * x; }

// Build one 16x32 bf16 WMMA fragment from LDS. CDNA5 16-bit A/B layout:
// lanes 0-15 hold K = {base+0..7, base+16..23}, lanes 16-31 hold K = {base+8..15, base+24..31}.
__device__ __forceinline__ bf16x16 load_frag16(const __bf16* p) {
  union { bf16x16 v; bf16x8 h[2]; } u;
  u.h[0] = *(const bf16x8*)(p);
  u.h[1] = *(const bf16x8*)(p + 16);
  return u.v;
}

// ---------------------------------------------------------------------------
// h0 = leaky([node_feat[src] | edge_feat] @ W_edge)   K = 80 padded to 96
// ---------------------------------------------------------------------------
__global__ __launch_bounds__(256) void k_edge_init(
    const float* __restrict__ nodef, const float* __restrict__ edgef,
    const int* __restrict__ src, const float* __restrict__ Wedge,
    __bf16* __restrict__ h0, int numBlk)
{
  __shared__ __align__(16) __bf16 Wt[128][96];      // W_edge transposed, bf16
  __shared__ __align__(16) __bf16 As[8][16][96];    // per-wave A staging
  const int tid = threadIdx.x;
  for (int idx = tid; idx < 128 * 96; idx += 256) {
    int j = idx / 96, k = idx - j * 96;
    Wt[j][k] = (k < 80) ? (__bf16)Wedge[k * 128 + j] : (__bf16)0.f;
  }
  __syncthreads();
  const int lane = tid & 31, wid = tid >> 5;
  const int half = lane >> 4, l15 = lane & 15;
  for (int idx = lane; idx < 16 * 16; idx += 32)    // zero K-pad once
    As[wid][idx >> 4][80 + (idx & 15)] = (__bf16)0.f;
  const int gw = blockIdx.x * 8 + wid, nw = gridDim.x * 8;
  for (int blk = gw; blk < numBlk; blk += nw) {
    for (int r = 0; r < 16; ++r) {
      const size_t e = (size_t)blk * 16 + r;
      if (lane < 16) {
        const int s = src[e];
        float4 a = *(const float4*)(nodef + (size_t)s * 64 + lane * 4);
        bf16x4 m; m[0]=(__bf16)a.x; m[1]=(__bf16)a.y; m[2]=(__bf16)a.z; m[3]=(__bf16)a.w;
        *(bf16x4*)&As[wid][r][lane * 4] = m;
      } else if (lane < 20) {
        float4 a = *(const float4*)(edgef + e * 16 + (lane - 16) * 4);
        bf16x4 m; m[0]=(__bf16)a.x; m[1]=(__bf16)a.y; m[2]=(__bf16)a.z; m[3]=(__bf16)a.w;
        *(bf16x4*)&As[wid][r][64 + (lane - 16) * 4] = m;
      }
    }
    bf16x16 af[3];
#pragma unroll
    for (int kt = 0; kt < 3; ++kt)
      af[kt] = load_frag16(&As[wid][l15][kt * 32 + half * 8]);
#pragma unroll
    for (int nt = 0; nt < 8; ++nt) {
      f32x8 acc = {};
#pragma unroll
      for (int kt = 0; kt < 3; ++kt) {
        bf16x16 bf = load_frag16(&Wt[nt * 16 + l15][kt * 32 + half * 8]);
        acc = __builtin_amdgcn_wmma_f32_16x16x32_bf16(false, af[kt], false, bf,
                                                      (short)0, acc, false, false);
      }
      const int n = nt * 16 + l15, mb = half * 8;
#pragma unroll
      for (int r2 = 0; r2 < 8; ++r2) {
        const size_t row = (size_t)blk * 16 + mb + r2;
        h0[row * 128 + n] = (__bf16)leakyf(acc[r2]);
      }
    }
  }
}

// ---------------------------------------------------------------------------
// h_new = leaky((sum0[src] - h_old[rev]) @ W_upd + h0)     K = 128
// ---------------------------------------------------------------------------
__global__ __launch_bounds__(256) void k_msg(
    const float* __restrict__ sum0, const __bf16* __restrict__ hold,
    const __bf16* __restrict__ h0, const int* __restrict__ src,
    const int* __restrict__ rev, const float* __restrict__ Wupd,
    __bf16* __restrict__ hnew, int numBlk)
{
  __shared__ __align__(16) __bf16 Wt[128][128];
  __shared__ __align__(16) __bf16 As[8][16][128];
  const int tid = threadIdx.x;
  for (int idx = tid; idx < 128 * 128; idx += 256) {
    int j = idx >> 7, k = idx & 127;
    Wt[j][k] = (__bf16)Wupd[k * 128 + j];
  }
  __syncthreads();
  const int lane = tid & 31, wid = tid >> 5;
  const int half = lane >> 4, l15 = lane & 15;
  const int gw = blockIdx.x * 8 + wid, nw = gridDim.x * 8;
  for (int blk = gw; blk < numBlk; blk += nw) {
    for (int r = 0; r < 16; ++r) {
      const size_t e = (size_t)blk * 16 + r;
      const int s = src[e], rv = rev[e];                 // uniform -> s_load
      float4 a = *(const float4*)(sum0 + (size_t)s * 128 + lane * 4);
      bf16x4 hb = *(const bf16x4*)(hold + (size_t)rv * 128 + lane * 4);
      bf16x4 m;
      m[0] = (__bf16)(a.x - (float)hb[0]);
      m[1] = (__bf16)(a.y - (float)hb[1]);
      m[2] = (__bf16)(a.z - (float)hb[2]);
      m[3] = (__bf16)(a.w - (float)hb[3]);
      *(bf16x4*)&As[wid][r][lane * 4] = m;
    }
    bf16x16 af[4];
#pragma unroll
    for (int kt = 0; kt < 4; ++kt)
      af[kt] = load_frag16(&As[wid][l15][kt * 32 + half * 8]);
#pragma unroll
    for (int nt = 0; nt < 8; ++nt) {
      f32x8 acc = {};
#pragma unroll
      for (int kt = 0; kt < 4; ++kt) {
        bf16x16 bf = load_frag16(&Wt[nt * 16 + l15][kt * 32 + half * 8]);
        acc = __builtin_amdgcn_wmma_f32_16x16x32_bf16(false, af[kt], false, bf,
                                                      (short)0, acc, false, false);
      }
      const int n = nt * 16 + l15, mb = half * 8;
#pragma unroll
      for (int r2 = 0; r2 < 8; ++r2) {
        const size_t row = (size_t)blk * 16 + mb + r2;
        float v = acc[r2] + (float)h0[row * 128 + n];
        hnew[row * 128 + n] = (__bf16)leakyf(v);
      }
    }
  }
}

// ---------------------------------------------------------------------------
// hn = leaky([node_feat | m_node] @ W_node + b_node)   K = 192
// ---------------------------------------------------------------------------
__global__ __launch_bounds__(256) void k_node(
    const float* __restrict__ nodef, const float* __restrict__ mnode,
    const float* __restrict__ Wnode, const float* __restrict__ bnode,
    float* __restrict__ hn, int numBlk)
{
  __shared__ __align__(16) __bf16 Wt[128][192];
  __shared__ __align__(16) __bf16 As[8][16][192];
  __shared__ float bias[128];
  const int tid = threadIdx.x;
  for (int idx = tid; idx < 128 * 192; idx += 256) {
    int j = idx / 192, k = idx - j * 192;
    Wt[j][k] = (__bf16)Wnode[k * 128 + j];
  }
  if (tid < 128) bias[tid] = bnode[tid];
  __syncthreads();
  const int lane = tid & 31, wid = tid >> 5;
  const int half = lane >> 4, l15 = lane & 15;
  const int gw = blockIdx.x * 8 + wid, nw = gridDim.x * 8;
  for (int blk = gw; blk < numBlk; blk += nw) {
    for (int r = 0; r < 16; ++r) {
      const size_t nidx = (size_t)blk * 16 + r;
      if (lane < 16) {
        float4 a = *(const float4*)(nodef + nidx * 64 + lane * 4);
        bf16x4 m; m[0]=(__bf16)a.x; m[1]=(__bf16)a.y; m[2]=(__bf16)a.z; m[3]=(__bf16)a.w;
        *(bf16x4*)&As[wid][r][lane * 4] = m;
      }
      float4 b = *(const float4*)(mnode + nidx * 128 + lane * 4);
      bf16x4 mm; mm[0]=(__bf16)b.x; mm[1]=(__bf16)b.y; mm[2]=(__bf16)b.z; mm[3]=(__bf16)b.w;
      *(bf16x4*)&As[wid][r][64 + lane * 4] = mm;
    }
    bf16x16 af[6];
#pragma unroll
    for (int kt = 0; kt < 6; ++kt)
      af[kt] = load_frag16(&As[wid][l15][kt * 32 + half * 8]);
#pragma unroll
    for (int nt = 0; nt < 8; ++nt) {
      f32x8 acc = {};
#pragma unroll
      for (int kt = 0; kt < 6; ++kt) {
        bf16x16 bf = load_frag16(&Wt[nt * 16 + l15][kt * 32 + half * 8]);
        acc = __builtin_amdgcn_wmma_f32_16x16x32_bf16(false, af[kt], false, bf,
                                                      (short)0, acc, false, false);
      }
      const int n = nt * 16 + l15, mb = half * 8;
#pragma unroll
      for (int r2 = 0; r2 < 8; ++r2) {
        const size_t row = (size_t)blk * 16 + mb + r2;
        hn[row * 128 + n] = leakyf(acc[r2] + bias[n]);
      }
    }
  }
}

// ---------------------------------------------------------------------------
// Segment sums via hardware f32 atomics (accumulator is L2-resident)
// ---------------------------------------------------------------------------
__global__ __launch_bounds__(256) void k_scatter_edges(
    const __bf16* __restrict__ h, const int* __restrict__ dst,
    float* __restrict__ sum0, int E)
{
  const long tid = (long)blockIdx.x * blockDim.x + threadIdx.x;
  const long e = tid >> 5;
  if (e >= E) return;
  const int f = (int)(tid & 31) * 4;
  const int d = dst[e];
  bf16x4 hv = *(const bf16x4*)(h + e * 128 + f);
  float* sp = sum0 + (size_t)d * 128 + f;
  unsafeAtomicAdd(sp + 0, (float)hv[0]);
  unsafeAtomicAdd(sp + 1, (float)hv[1]);
  unsafeAtomicAdd(sp + 2, (float)hv[2]);
  unsafeAtomicAdd(sp + 3, (float)hv[3]);
}

__global__ __launch_bounds__(256) void k_scatter_nodes(
    const float* __restrict__ hn, const int* __restrict__ gid,
    float* __restrict__ gf, int N)
{
  const long tid = (long)blockIdx.x * blockDim.x + threadIdx.x;
  const long n = tid >> 5;
  if (n >= N) return;
  const int f = (int)(tid & 31) * 4;
  const int g = gid[n];
  const float4 v = *(const float4*)(hn + n * 128 + f);
  float* sp = gf + (size_t)g * 128 + f;
  unsafeAtomicAdd(sp + 0, v.x);
  unsafeAtomicAdd(sp + 1, v.y);
  unsafeAtomicAdd(sp + 2, v.z);
  unsafeAtomicAdd(sp + 3, v.w);
}

// ---------------------------------------------------------------------------
// Tiny MoE tail (B = 256 rows; <1% of total work)
// ---------------------------------------------------------------------------
__global__ void k_concat(const float* __restrict__ gf, const float* __restrict__ extra,
                         float* __restrict__ x, int B)
{
  int t = blockIdx.x * blockDim.x + threadIdx.x;
  if (t >= B * 144) return;
  int b = t / 144, j = t - b * 144;
  x[t] = (j < 128) ? gf[b * 128 + j] : extra[b * 16 + (j - 128)];
}

__global__ void k_small_gemm(const float* __restrict__ A, const float* __restrict__ W,
                             const float* __restrict__ bias, float* __restrict__ out,
                             int M, int K, int Nout, int doLeaky)
{
  int t = blockIdx.x * blockDim.x + threadIdx.x;
  if (t >= M * Nout) return;
  int b = t / Nout, j = t - b * Nout;
  float acc = bias[j];
  const float* a = A + (size_t)b * K;
  for (int k = 0; k < K; ++k) acc += a[k] * W[(size_t)k * Nout + j];
  out[t] = doLeaky ? leakyf(acc) : acc;
}

__global__ void k_expert1(const float* __restrict__ X, const float* __restrict__ W,
                          const float* __restrict__ bias, float* __restrict__ out, int B)
{
  int t = blockIdx.x * blockDim.x + threadIdx.x;
  if (t >= B * 8 * 128) return;
  int hh = t & 127, e = (t >> 7) & 7, b = t >> 10;
  float acc = bias[e * 128 + hh];
  const float* x = X + (size_t)b * 144;
  const float* w = W + ((size_t)e * 144) * 128 + hh;
  for (int k = 0; k < 144; ++k) acc += x[k] * w[(size_t)k * 128];
  out[t] = leakyf(acc);
}

__global__ void k_expert2(const float* __restrict__ E1, const float* __restrict__ W,
                          const float* __restrict__ bias, float* __restrict__ out, int B)
{
  int t = blockIdx.x * blockDim.x + threadIdx.x;
  if (t >= B * 8 * 128) return;
  int hh = t & 127, e = (t >> 7) & 7, b = t >> 10;
  float acc = bias[e * 128 + hh];
  const float* x = E1 + ((size_t)b * 8 + e) * 128;
  const float* w = W + ((size_t)e * 128) * 128 + hh;
  for (int k = 0; k < 128; ++k) acc += x[k] * w[(size_t)k * 128];
  out[t] = leakyf(acc);
}

__global__ void k_final(const float* __restrict__ E2buf, const float* __restrict__ EW3,
                        const float* __restrict__ Eb3, const float* __restrict__ logits,
                        float* __restrict__ out, int B)
{
  int b = blockIdx.x * blockDim.x + threadIdx.x;
  if (b >= B) return;
  float ev[8], lgv[8], mx = -3.0e38f;
  for (int e = 0; e < 8; ++e) {
    float s = Eb3[e];
    const float* p = E2buf + ((size_t)b * 8 + e) * 128;
    const float* w = EW3 + (size_t)e * 128;
    for (int h = 0; h < 128; ++h) s += p[h] * w[h];
    ev[e] = s;
    lgv[e] = logits[b * 8 + e];
    mx = fmaxf(mx, lgv[e]);
  }
  float den = 0.f, num = 0.f;
  for (int e = 0; e < 8; ++e) {
    float pe = __expf(lgv[e] - mx);
    den += pe; num += pe * ev[e];
  }
  out[b] = num / den;
}

// ---------------------------------------------------------------------------
extern "C" void kernel_launch(void* const* d_in, const int* in_sizes, int n_in,
                              void* d_out, int out_size, void* d_ws, size_t ws_size,
                              hipStream_t stream) {
  const float* node_feat = (const float*)d_in[0];
  const float* edge_feat = (const float*)d_in[1];
  const int*   src  = (const int*)d_in[2];
  const int*   dst  = (const int*)d_in[3];
  const int*   rev  = (const int*)d_in[4];
  const int*   gid  = (const int*)d_in[5];
  const float* extra = (const float*)d_in[6];
  const float* W_edge = (const float*)d_in[7];
  const float* W_upd  = (const float*)d_in[8];
  const float* W_node = (const float*)d_in[9];
  const float* b_node = (const float*)d_in[10];
  const float* Wg1 = (const float*)d_in[11];
  const float* bg1 = (const float*)d_in[12];
  const float* Wg2 = (const float*)d_in[13];
  const float* bg2 = (const float*)d_in[14];
  const float* Wg3 = (const float*)d_in[15];
  const float* bg3 = (const float*)d_in[16];
  const float* EW1 = (const float*)d_in[17];
  const float* Eb1 = (const float*)d_in[18];
  const float* EW2 = (const float*)d_in[19];
  const float* Eb2 = (const float*)d_in[20];
  const float* EW3 = (const float*)d_in[21];
  const float* Eb3 = (const float*)d_in[22];
  float* out = (float*)d_out;

  const int E = in_sizes[2];           // 640000
  const int N = in_sizes[0] / 64;      // 50000
  const int B = in_sizes[6] / 16;      // 256
  const int EB = E / 16;
  const int NB = N / 16;

  char* wptr = (char*)d_ws;
  auto alloc = [&](size_t bytes) -> char* {
    char* p = wptr;
    wptr += (bytes + 255) & ~(size_t)255;
    return p;
  };
  __bf16* h0buf = (__bf16*)alloc((size_t)E * 128 * sizeof(__bf16));
  __bf16* hA    = (__bf16*)alloc((size_t)E * 128 * sizeof(__bf16));
  __bf16* hB    = (__bf16*)alloc((size_t)E * 128 * sizeof(__bf16));
  float* sum0 = (float*)alloc((size_t)N * 128 * sizeof(float));
  float* hn   = (float*)alloc((size_t)N * 128 * sizeof(float));
  float* gf   = (float*)alloc((size_t)B * 128 * sizeof(float));
  float* xbuf = (float*)alloc((size_t)B * 144 * sizeof(float));
  float* g1   = (float*)alloc((size_t)B * 128 * sizeof(float));
  float* g2   = (float*)alloc((size_t)B * 128 * sizeof(float));
  float* lg   = (float*)alloc((size_t)B * 8 * sizeof(float));
  float* e1   = (float*)alloc((size_t)B * 8 * 128 * sizeof(float));
  float* e2   = (float*)alloc((size_t)B * 8 * 128 * sizeof(float));

  const dim3 blk(256);

  // h0 (also serves as h for round 1)
  k_edge_init<<<1024, blk, 0, stream>>>(node_feat, edge_feat, src, W_edge, h0buf, EB);

  // 4 message-passing rounds, rotating buffers h0 -> A -> B -> A -> B
  const __bf16* hold = h0buf;
  __bf16* bufs[2] = {hA, hB};
  const int segEdgeBlocks = (E * 32 + 255) / 256;
  for (int r = 0; r < 4; ++r) {
    hipMemsetAsync(sum0, 0, (size_t)N * 128 * sizeof(float), stream);
    k_scatter_edges<<<segEdgeBlocks, blk, 0, stream>>>(hold, dst, sum0, E);
    __bf16* hnew = bufs[r & 1];
    k_msg<<<1024, blk, 0, stream>>>(sum0, hold, h0buf, src, rev, W_upd, hnew, EB);
    hold = hnew;
  }

  // node readout
  hipMemsetAsync(sum0, 0, (size_t)N * 128 * sizeof(float), stream);
  k_scatter_edges<<<segEdgeBlocks, blk, 0, stream>>>(hold, dst, sum0, E);
  k_node<<<400, blk, 0, stream>>>(node_feat, sum0, W_node, b_node, hn, NB);

  hipMemsetAsync(gf, 0, (size_t)B * 128 * sizeof(float), stream);
  k_scatter_nodes<<<(N * 32 + 255) / 256, blk, 0, stream>>>(hn, gid, gf, N);
  k_concat<<<(B * 144 + 255) / 256, blk, 0, stream>>>(gf, extra, xbuf, B);

  // gating
  k_small_gemm<<<(B * 128 + 255) / 256, blk, 0, stream>>>(xbuf, Wg1, bg1, g1, B, 144, 128, 1);
  k_small_gemm<<<(B * 128 + 255) / 256, blk, 0, stream>>>(g1, Wg2, bg2, g2, B, 128, 128, 1);
  k_small_gemm<<<(B * 8 + 255) / 256, blk, 0, stream>>>(g2, Wg3, bg3, lg, B, 128, 8, 0);

  // experts + softmax-weighted combine
  k_expert1<<<(B * 8 * 128 + 255) / 256, blk, 0, stream>>>(xbuf, EW1, Eb1, e1, B);
  k_expert2<<<(B * 8 * 128 + 255) / 256, blk, 0, stream>>>(e1, EW2, Eb2, e2, B);
  k_final<<<(B + 255) / 256, blk, 0, stream>>>(e2, EW3, Eb3, lg, out, B);
}
